// GMBLayer_78503412236538
// MI455X (gfx1250) — matrix operand order repaired and checked by reading, not compiled, not run on hardware
//
#include <hip/hip_runtime.h>
#include <hip/hip_bf16.h>

// ---------------- problem constants (match reference) ----------------
#define HDIM   128
#define NNODES 16384
#define NEDGES 262144
#define KDIST  16
#define NPER   512
#define BGRAPH 32
#define DSTATE 16
#define DTRANK 8

typedef __bf16 v16bf __attribute__((ext_vector_type(16)));
typedef float  v8f   __attribute__((ext_vector_type(8)));

union FragBF { v16bf v; __bf16 e[16]; uint4 q[2]; };
union Pack4  { __bf16 h[4]; uint2 u; };

// ---------------- helpers ----------------
// block reduce for 128-thread blocks: wave32 shfl + 4-slot shared combine
__device__ __forceinline__ float block_sum_128(float v, float* sb4) {
    for (int m = 16; m > 0; m >>= 1) v += __shfl_xor(v, m, 32);
    int wave = threadIdx.x >> 5;
    if ((threadIdx.x & 31) == 0) sb4[wave] = v;
    __syncthreads();
    float r = sb4[0] + sb4[1] + sb4[2] + sb4[3];
    __syncthreads();
    return r;
}
__device__ __forceinline__ float sigm(float x) { return 1.f / (1.f + __expf(-x)); }

// ---------------- generic bf16-WMMA GEMM (all shapes tile-aligned) ----------------
// C[m,n] = act( sum_k A[m,k]*B[k,n] + bias[n] ) + resid[m,n]
// bT=1: B given as W[n,k] (row-major [Nd,Kd]);  bT=0: B given as [Kd,Nd].
// Requires: M%64==0, Nd%64==0, Kd%32==0, all row strides %4==0, 16B-aligned bases.
// einsum=1: batched mask-GEMM; z = b*16+kd, Aoff=z*sA, Boff=b*sB, Coff=kd*sC+b*sB.
// act: 0=none 1=relu 2=gelu(erf) 3=softplus
__global__ __launch_bounds__(128)
void k_gemm(const float* __restrict__ A, long lda, long sA,
            const float* __restrict__ Bm, long ldb, long sB, int bT,
            const float* __restrict__ bias,
            const float* __restrict__ resid, long ldr,
            float* __restrict__ C, long ldc, long sC,
            int M, int Nd, int Kd, int act, int einsum)
{
    __shared__ __align__(16) __bf16 As[2][64][40];
    __shared__ __align__(16) __bf16 Bs[2][64][40];

    int z = blockIdx.z;
    long Aoff, Boff, Coff;
    if (einsum) {
        int b = z >> 4, kd = z & 15;
        Aoff = (long)z  * sA;
        Boff = (long)b  * sB;
        Coff = (long)kd * sC + (long)b * sB;
    } else {
        Aoff = (long)z * sA; Boff = (long)z * sB; Coff = (long)z * sC;
    }
    int m0 = blockIdx.x * 64, n0 = blockIdx.y * 64;
    int tid = threadIdx.x, lane = tid & 31, wave = tid >> 5;
    int g = lane >> 4;

    // gather all global loads first (8 b128 loads in flight), then convert+commit
    auto stage = [&](int buf, int k0) {
        float4 fA[4], fB[4];
#pragma unroll
        for (int i = 0; i < 4; ++i) {
            int q = i * 128 + tid;
            int r = q >> 3, cb = (q & 7) * 4;
            fA[i] = *(const float4*)(A + Aoff + (long)(m0 + r) * lda + (k0 + cb));
        }
        if (bT) {
#pragma unroll
            for (int i = 0; i < 4; ++i) {
                int q = i * 128 + tid;
                int n = q >> 3, cb = (q & 7) * 4;
                fB[i] = *(const float4*)(Bm + Boff + (long)(n0 + n) * ldb + (k0 + cb));
            }
        } else {
#pragma unroll
            for (int i = 0; i < 4; ++i) {
                int q = i * 128 + tid;
                int kk = q >> 4, nb = (q & 15) * 4;
                fB[i] = *(const float4*)(Bm + Boff + (long)(k0 + kk) * ldb + (n0 + nb));
            }
        }
#pragma unroll
        for (int i = 0; i < 4; ++i) {
            int q = i * 128 + tid;
            int r = q >> 3, cb = (q & 7) * 4;
            Pack4 p;
            p.h[0] = (__bf16)fA[i].x; p.h[1] = (__bf16)fA[i].y;
            p.h[2] = (__bf16)fA[i].z; p.h[3] = (__bf16)fA[i].w;
            *(uint2*)&As[buf][r][cb] = p.u;
        }
        if (bT) {
#pragma unroll
            for (int i = 0; i < 4; ++i) {
                int q = i * 128 + tid;
                int n = q >> 3, cb = (q & 7) * 4;
                Pack4 p;
                p.h[0] = (__bf16)fB[i].x; p.h[1] = (__bf16)fB[i].y;
                p.h[2] = (__bf16)fB[i].z; p.h[3] = (__bf16)fB[i].w;
                *(uint2*)&Bs[buf][n][cb] = p.u;
            }
        } else {
#pragma unroll
            for (int i = 0; i < 4; ++i) {
                int q = i * 128 + tid;
                int kk = q >> 4, nb = (q & 15) * 4;
                Bs[buf][nb + 0][kk] = (__bf16)fB[i].x;
                Bs[buf][nb + 1][kk] = (__bf16)fB[i].y;
                Bs[buf][nb + 2][kk] = (__bf16)fB[i].z;
                Bs[buf][nb + 3][kk] = (__bf16)fB[i].w;
            }
        }
    };

    v8f acc[4];
    v8f zero8 = {0.f,0.f,0.f,0.f,0.f,0.f,0.f,0.f};
    for (int s = 0; s < 4; ++s) acc[s] = zero8;

    int nt = Kd >> 5;
    stage(0, 0);
    for (int kt = 0; kt < nt; ++kt) {
        __syncthreads();
        int cur = kt & 1;
        if (kt + 1 < nt) stage(cur ^ 1, (kt + 1) * 32);   // overlap next tile with math

        // preload A fragment + all 4 B fragments, then issue WMMAs back-to-back
        int mr = wave * 16 + (lane & 15);
        FragBF fa, fb[4];
        fa.q[0] = *(const uint4*)&As[cur][mr][g * 8];
        fa.q[1] = *(const uint4*)&As[cur][mr][16 + g * 8];
#pragma unroll
        for (int s = 0; s < 4; ++s) {
            int nl = s * 16 + (lane & 15);
            fb[s].q[0] = *(const uint4*)&Bs[cur][nl][g * 8];
            fb[s].q[1] = *(const uint4*)&Bs[cur][nl][16 + g * 8];
        }
#pragma unroll
        for (int s = 0; s < 4; ++s)
            acc[s] = __builtin_amdgcn_wmma_f32_16x16x32_bf16(
                         false, fa.v, false, fb[s].v, (short)0, acc[s], false, false);
    }

    // epilogue: lanes 0-15 -> M=r, lanes 16-31 -> M=8+r, N=lane&15 (ISA C/D layout)
    for (int s = 0; s < 4; ++s) {
        int n = n0 + s * 16 + (lane & 15);
        float bv = bias ? bias[n] : 0.f;
        for (int r = 0; r < 8; ++r) {
            int m = m0 + wave * 16 + g * 8 + r;
            float v = acc[s][r] + bv;
            if (act == 1)      v = fmaxf(v, 0.f);
            else if (act == 2) v = 0.5f * v * (1.f + erff(v * 0.70710678118f));
            else if (act == 3) v = (v > 20.f) ? v : log1pf(__expf(v));
            if (resid) v += resid[(long)m * ldr + n];
            C[Coff + (long)m * ldc + n] = v;
        }
    }
}

// ---------------- zero-pad small weights so every GEMM is tile-aligned ----------------
__global__ __launch_bounds__(256)
void k_pad(const float* __restrict__ xpj, const float* __restrict__ dtw,
           float* __restrict__ xpjP, float* __restrict__ dtwP)
{
    int i = blockIdx.x * 256 + threadIdx.x;
    if (i < 64 * 128) {           // x_proj [40,128] -> [64,128]
        int r = i >> 7, c = i & 127;
        xpjP[i] = (r < 40) ? xpj[r * 128 + c] : 0.f;
    }
    if (i < 128 * 32) {           // dt_w [128,8] -> [128,32]
        int r = i >> 5, c = i & 31;
        dtwP[i] = (c < 8) ? dtw[r * 8 + c] : 0.f;
    }
}

// ---------------- edge gating + atomic segment sums ----------------
__global__ __launch_bounds__(128)
void k_edge(const int* __restrict__ ei, const float* __restrict__ pe,
            const float* __restrict__ Bx, const float* __restrict__ Dx,
            const float* __restrict__ Ex, const float* __restrict__ Ce,
            const float* __restrict__ rw1, const float* __restrict__ rb1,
            const float* __restrict__ rw2, const float* __restrict__ rb2,
            float* __restrict__ num, float* __restrict__ den)
{
    __shared__ float sb[4];
    long e = blockIdx.x; int h = threadIdx.x;
    int s = ei[e], d = ei[(long)NEDGES + e];
    float rr = 0.f;
    for (int k = 0; k < 8; ++k) {
        float df = pe[(long)d * 8 + k] - pe[(long)s * 8 + k];
        rr += df * df;
    }
    float hv  = fmaxf(rr * rw1[h] + rb1[h], 0.f);
    float tot = block_sum_128(hv * rw2[h], sb);
    float rsc = sigm(tot + rb2[0]);
    float eij = Dx[(long)d * HDIM + h] + Ex[(long)s * HDIM + h] + Ce[e * HDIM + h];
    float sg  = sigm(eij) * rsc;
    atomicAdd(&num[(long)d * HDIM + h], sg * Bx[(long)s * HDIM + h]);
    atomicAdd(&den[(long)d * HDIM + h], sg);
}

// ---------------- BatchNorm pass 1: t = Ax + num/(den+eps), channel stats ----------------
__global__ __launch_bounds__(128)
void k_bn_stats(const float* __restrict__ Ax, const float* __restrict__ num,
                const float* __restrict__ den, float* __restrict__ tbuf,
                float* __restrict__ bsum, float* __restrict__ bsq)
{
    int h = threadIdx.x;
    long r0 = (long)blockIdx.x * 128;
    float s = 0.f, s2 = 0.f;
    for (int i = 0; i < 128; ++i) {
        long idx = (r0 + i) * HDIM + h;
        float v = Ax[idx] + num[idx] / (den[idx] + 1e-6f);
        tbuf[idx] = v; s += v; s2 += v * v;
    }
    atomicAdd(&bsum[h], s); atomicAdd(&bsq[h], s2);
}

// ---------------- BN apply + relu + double residual + LayerNorm -> local ----------------
__global__ __launch_bounds__(128)
void k_bn_local(const float* __restrict__ x, const float* __restrict__ tbuf,
                const float* __restrict__ bsum, const float* __restrict__ bsq,
                const float* __restrict__ bg, const float* __restrict__ bb,
                const float* __restrict__ lg, const float* __restrict__ lb,
                float* __restrict__ local)
{
    __shared__ float sb[4];
    int h = threadIdx.x; long row = blockIdx.x;
    float mu  = bsum[h] * (1.f / NNODES);
    float var = bsq[h] * (1.f / NNODES) - mu * mu;
    float v = (tbuf[row * HDIM + h] - mu) * rsqrtf(var + 1e-5f) * bg[h] + bb[h];
    v = fmaxf(v, 0.f);
    float pre = 2.f * x[row * HDIM + h] + v;
    float m  = block_sum_128(pre, sb) * (1.f / HDIM);
    float dd = pre - m;
    float vv = block_sum_128(dd * dd, sb) * (1.f / HDIM);
    local[row * HDIM + h] = dd * rsqrtf(vv + 1e-5f) * lg[h] + lb[h];
}

// ---------------- generic row LayerNorm ----------------
__global__ __launch_bounds__(128)
void k_rowln(const float* __restrict__ in, float* __restrict__ out,
             const float* __restrict__ g, const float* __restrict__ b)
{
    __shared__ float sb[4];
    int h = threadIdx.x; long row = blockIdx.x;
    float v  = in[row * HDIM + h];
    float m  = block_sum_128(v, sb) * (1.f / HDIM);
    float d  = v - m;
    float vv = block_sum_128(d * d, sb) * (1.f / HDIM);
    out[row * HDIM + h] = d * rsqrtf(vv + 1e-5f) * g[h] + b[h];
}

// ---------------- flip + LN_pre: seqln[n*16+j] = LN(xK[15-j, n]) ----------------
__global__ __launch_bounds__(128)
void k_seq_ln(const float* __restrict__ xK, float* __restrict__ seqln,
              const float* __restrict__ g, const float* __restrict__ b)
{
    __shared__ float sb[4];
    int h = threadIdx.x; long row = blockIdx.x;
    long n = row >> 4; int j = (int)(row & 15); int kd = 15 - j;
    float v  = xK[((long)kd * NNODES + n) * HDIM + h];
    float m  = block_sum_128(v, sb) * (1.f / HDIM);
    float d  = v - m;
    float vv = block_sum_128(d * d, sb) * (1.f / HDIM);
    seqln[row * HDIM + h] = d * rsqrtf(vv + 1e-5f) * g[h] + b[h];
}

// ---------------- causal depthwise conv (D_CONV=4) + SiLU ----------------
__global__ __launch_bounds__(256)
void k_conv(const float* __restrict__ xm, const float* __restrict__ cw,
            const float* __restrict__ cb, float* __restrict__ xc)
{
    long idx = (long)blockIdx.x * 256 + threadIdx.x;
    if (idx >= (long)NNODES * HDIM) return;
    long n = idx >> 7; int h = (int)(idx & 127);
    float w0 = cw[h * 4], w1 = cw[h * 4 + 1], w2 = cw[h * 4 + 2], w3 = cw[h * 4 + 3];
    float b = cb[h];
    float p3 = 0.f, p2 = 0.f, p1 = 0.f;
    for (int j = 0; j < KDIST; ++j) {
        float cur = xm[((n << 4) + j) * HDIM + h];
        float a = w0 * p3 + w1 * p2 + w2 * p1 + w3 * cur + b;
        xc[((n << 4) + j) * HDIM + h] = a * sigm(a);
        p3 = p2; p2 = p1; p1 = cur;
    }
}

// ---------------- selective scan; emit only last step, fused Dp & silu(z) ----------------
__global__ __launch_bounds__(256)
void k_scan(const float* __restrict__ dtf, const float* __restrict__ dtbc,
            const float* __restrict__ xc, const float* __restrict__ zl,
            const float* __restrict__ alog, const float* __restrict__ Dp,
            float* __restrict__ yp)
{
    long idx = (long)blockIdx.x * 256 + threadIdx.x;
    if (idx >= (long)NNODES * HDIM) return;
    long n = idx >> 7; int h = (int)(idx & 127);
    float a[DSTATE], st[DSTATE];
    for (int s = 0; s < DSTATE; ++s) { a[s] = -__expf(alog[h * DSTATE + s]); st[s] = 0.f; }
    float y = 0.f, xlast = 0.f;
    for (int j = 0; j < KDIST; ++j) {
        long base = (n << 4) + j;
        float dtv = dtf[base * HDIM + h];
        float xv  = xc[base * HDIM + h];
        const float* bc = dtbc + base * 64;        // padded stride 64
        for (int s = 0; s < DSTATE; ++s)
            st[s] = st[s] * __expf(dtv * a[s]) + dtv * xv * bc[8 + s];
        if (j == KDIST - 1) {
            for (int s = 0; s < DSTATE; ++s) y += st[s] * bc[24 + s];
            xlast = xv;
        }
    }
    y += xlast * Dp[h];
    float z = zl[n * HDIM + h];
    yp[n * HDIM + h] = y * (z * sigm(z));
}

// ---------------- elementwise exact GELU ----------------
__global__ __launch_bounds__(256)
void k_gelu(const float* __restrict__ in, float* __restrict__ out, long nelem)
{
    long idx = (long)blockIdx.x * 256 + threadIdx.x;
    if (idx >= nelem) return;
    float v = in[idx];
    out[idx] = 0.5f * v * (1.f + erff(v * 0.70710678118f));
}

// ---------------- final GLU + skips + LayerNorm -> d_out ----------------
__global__ __launch_bounds__(128)
void k_final(const float* __restrict__ t1, const float* __restrict__ t2,
             const float* __restrict__ skip0, const float* __restrict__ local,
             const float* __restrict__ g, const float* __restrict__ b,
             float* __restrict__ out)
{
    __shared__ float sb[4];
    int h = threadIdx.x; long row = blockIdx.x;
    long i = row * HDIM + h;
    float branch = t1[i] * sigm(t2[i]) + skip0[i];
    float pre = local[i] + branch;
    float m  = block_sum_128(pre, sb) * (1.f / HDIM);
    float d  = pre - m;
    float vv = block_sum_128(d * d, sb) * (1.f / HDIM);
    out[i] = d * rsqrtf(vv + 1e-5f) * g[h] + b[h];
}

// ---------------- host ----------------
static void launch_gemm(hipStream_t st, const float* A, long lda, long sA,
                        const float* B, long ldb, long sB, int bT,
                        const float* bias, const float* resid, long ldr,
                        float* C, long ldc, long sC,
                        int M, int Nd, int Kd, int act, int einsum, int batch)
{
    dim3 grid((M + 63) / 64, (Nd + 63) / 64, batch);
    k_gemm<<<grid, dim3(128), 0, st>>>(A, lda, sA, B, ldb, sB, bT, bias, resid, ldr,
                                       C, ldc, sC, M, Nd, Kd, act, einsum);
}

extern "C" void kernel_launch(void* const* d_in, const int* in_sizes, int n_in,
                              void* d_out, int out_size, void* d_ws, size_t ws_size,
                              hipStream_t stream)
{
    const float* x   = (const float*)d_in[0];
    const float* ea  = (const float*)d_in[1];
    const float* pe  = (const float*)d_in[2];
    const int*   ei  = (const int*)d_in[3];
    const float* dm  = (const float*)d_in[5];
    const float* gAw = (const float*)d_in[6],  *gAb = (const float*)d_in[7];
    const float* gBw = (const float*)d_in[8],  *gBb = (const float*)d_in[9];
    const float* gCw = (const float*)d_in[10], *gCb = (const float*)d_in[11];
    const float* gDw = (const float*)d_in[12], *gDb = (const float*)d_in[13];
    const float* gEw = (const float*)d_in[14], *gEb = (const float*)d_in[15];
    const float* rw1 = (const float*)d_in[16], *rb1 = (const float*)d_in[17];
    const float* rw2 = (const float*)d_in[18], *rb2 = (const float*)d_in[19];
    const float* bxg = (const float*)d_in[20], *bxb = (const float*)d_in[21];
    const float* llg = (const float*)d_in[24], *llb = (const float*)d_in[25];
    const float* m1lg= (const float*)d_in[26], *m1lb= (const float*)d_in[27];
    const float* m1w1= (const float*)d_in[28], *m1b1= (const float*)d_in[29];
    const float* m1w2= (const float*)d_in[30], *m1b2= (const float*)d_in[31];
    const float* lpg = (const float*)d_in[32], *lpb = (const float*)d_in[33];
    const float* inpj= (const float*)d_in[34];
    const float* cw  = (const float*)d_in[35], *cb  = (const float*)d_in[36];
    const float* xpj = (const float*)d_in[37];
    const float* dtw = (const float*)d_in[38], *dtb = (const float*)d_in[39];
    const float* alog= (const float*)d_in[40], *Dpw = (const float*)d_in[41];
    const float* opj = (const float*)d_in[42];
    const float* m2w1= (const float*)d_in[43], *m2b1= (const float*)d_in[44];
    const float* m2w2= (const float*)d_in[45], *m2b2= (const float*)d_in[46];
    const float* logm= (const float*)d_in[47], *lob = (const float*)d_in[48];

    const long NH  = (long)NNODES * HDIM;       // 2,097,152
    const long NKH = NH * KDIST;                // 33,554,432  (== NEDGES*HDIM)

    float* ws = (float*)d_ws;
    float* R1 = ws;                 // Ce -> xK -> xm -> dt_full
    float* R2 = ws + NKH;           // Ax..t -> seqln
    float* R3 = ws + 2 * NKH;       // xc
    float* Ax   = R2 + 0 * NH;
    float* Bxb  = R2 + 1 * NH;
    float* Dx   = R2 + 2 * NH;
    float* Ex   = R2 + 3 * NH;
    float* numb = R2 + 4 * NH;
    float* denb = R2 + 5 * NH;
    float* tbuf = R2 + 6 * NH;
    float* Ce   = R1;
    float* xK   = R1;               // after edge phase
    float* xm   = R1;               // after seqln built
    float* dtf  = R1;               // after conv consumed xm
    float* seqln= R2;               // after BN/local done
    float* xc   = R3;

    float* sm    = ws + 3 * NKH;
    float* dtbc  = sm;                         // N*K*64 (padded stride)
    float* local = dtbc + (long)NNODES * KDIST * 64;
    float* skip0 = local + NH;
    float* zl    = skip0 + NH;
    float* ypre  = zl + NH;
    float* ylp   = ypre + NH;
    float* gbuf  = ylp + NH;
    float* t1    = gbuf + NH;
    float* t2    = t1 + NH;
    float* h1    = t2 + NH;
    float* bsum  = h1 + NH;        // 128
    float* bsq   = bsum + 128;     // 128
    float* xpjP  = bsq + 128;      // 64*128
    float* dtwP  = xpjP + 64 * 128;// 128*32

    k_pad<<<32, 256, 0, stream>>>(xpj, dtw, xpjP, dtwP);

    // ---- GatedGCN linears (bf16 WMMA) ----
    launch_gemm(stream, x, HDIM, 0, gAw, HDIM, 0, 1, gAb, nullptr, 0, Ax,  HDIM, 0, NNODES, HDIM, HDIM, 0, 0, 1);
    launch_gemm(stream, x, HDIM, 0, gBw, HDIM, 0, 1, gBb, nullptr, 0, Bxb, HDIM, 0, NNODES, HDIM, HDIM, 0, 0, 1);
    launch_gemm(stream, x, HDIM, 0, gDw, HDIM, 0, 1, gDb, nullptr, 0, Dx,  HDIM, 0, NNODES, HDIM, HDIM, 0, 0, 1);
    launch_gemm(stream, x, HDIM, 0, gEw, HDIM, 0, 1, gEb, nullptr, 0, Ex,  HDIM, 0, NNODES, HDIM, HDIM, 0, 0, 1);
    launch_gemm(stream, ea, HDIM, 0, gCw, HDIM, 0, 1, gCb, nullptr, 0, Ce, HDIM, 0, NEDGES, HDIM, HDIM, 0, 0, 1);

    hipMemsetAsync(numb, 0, 2 * NH * sizeof(float), stream);   // num + den (adjacent)
    hipMemsetAsync(bsum, 0, 256 * sizeof(float), stream);

    k_edge<<<NEDGES, 128, 0, stream>>>(ei, pe, Bxb, Dx, Ex, Ce, rw1, rb1, rw2, rb2, numb, denb);
    k_bn_stats<<<128, 128, 0, stream>>>(Ax, numb, denb, tbuf, bsum, bsq);
    k_bn_local<<<NNODES, 128, 0, stream>>>(x, tbuf, bsum, bsq, bxg, bxb, llg, llb, local);

    // ---- distance-mask einsum: 512 batched [512x128x512] GEMMs -> xK [K,N,H] ----
    launch_gemm(stream, dm, NPER, (long)NPER * NPER,
                x, HDIM, (long)NPER * HDIM, 0, nullptr, nullptr, 0,
                xK, HDIM, (long)NNODES * HDIM,
                NPER, HDIM, NPER, 0, 1, BGRAPH * KDIST);

    // ---- MLP1 (k=0 slice only) ----
    k_rowln<<<NNODES, 128, 0, stream>>>(xK, h1, m1lg, m1lb);
    launch_gemm(stream, h1, HDIM, 0, m1w1, HDIM, 0, 1, m1b1, nullptr, 0, t1, HDIM, 0, NNODES, HDIM, HDIM, 2, 0, 1);
    launch_gemm(stream, t1, HDIM, 0, m1w2, HDIM, 0, 1, m1b2, xK, HDIM, skip0, HDIM, 0, NNODES, HDIM, HDIM, 0, 0, 1);

    // ---- Mamba front-end ----
    k_seq_ln<<<NNODES * KDIST, 128, 0, stream>>>(xK, seqln, lpg, lpb);
    // z only needed at j=K-1 (rows n*16+15)
    launch_gemm(stream, seqln + (KDIST - 1) * HDIM, (long)KDIST * HDIM, 0,
                inpj + HDIM * HDIM, HDIM, 0, 1, nullptr, nullptr, 0,
                zl, HDIM, 0, NNODES, HDIM, HDIM, 0, 0, 1);
    launch_gemm(stream, seqln, HDIM, 0, inpj, HDIM, 0, 1, nullptr, nullptr, 0,
                xm, HDIM, 0, NNODES * KDIST, HDIM, HDIM, 0, 0, 1);
    k_conv<<<(unsigned)((NH + 255) / 256), 256, 0, stream>>>(xm, cw, cb, xc);
    launch_gemm(stream, xc, HDIM, 0, xpjP, HDIM, 0, 1, nullptr, nullptr, 0,
                dtbc, 64, 0, NNODES * KDIST, 64, HDIM, 0, 0, 1);
    launch_gemm(stream, dtbc, 64, 0, dtwP, 32, 0, 1, dtb, nullptr, 0,
                dtf, HDIM, 0, NNODES * KDIST, HDIM, 32, 3, 0, 1);

    // ---- scan + out_proj ----
    k_scan<<<(unsigned)((NH + 255) / 256), 256, 0, stream>>>(dtf, dtbc, xc, zl, alog, Dpw, ypre);
    launch_gemm(stream, ypre, HDIM, 0, opj, HDIM, 0, 1, nullptr, nullptr, 0,
                ylp, HDIM, 0, NNODES, HDIM, HDIM, 0, 0, 1);

    // ---- MLP2 (full GLU) + final LN ----
    k_gelu<<<(unsigned)((NH + 255) / 256), 256, 0, stream>>>(ylp, gbuf, NH);
    launch_gemm(stream, gbuf, HDIM, 0, m2w1, HDIM, 0, 1, m2b1, nullptr, 0, t1, HDIM, 0, NNODES, HDIM, HDIM, 0, 0, 1);
    launch_gemm(stream, gbuf, HDIM, 0, m2w2, HDIM, 0, 1, m2b2, nullptr, 0, t2, HDIM, 0, NNODES, HDIM, HDIM, 0, 0, 1);
    k_final<<<NNODES, 128, 0, stream>>>(t1, t2, skip0, local, logm, lob, (float*)d_out);
}
